// Sequence_32856499814488
// MI455X (gfx1250) — matrix operand order repaired and checked by reading, not compiled, not run on hardware
//
#include <hip/hip_runtime.h>
#include <hip/hip_bf16.h>
#include <cstdint>

// ---------------------------------------------------------------------------
// 2-layer LSTM (T=1024, B=128, H=1536) + final linear, for MI455X (gfx1250).
// Persistent-grid kernel, bf16 WMMA (v_wmma_f32_16x16x32_bf16):
//  - cell state in VGPRs for all 1024 steps
//  - W_hh0 fragments resident in LDS (192 KB, ds_load_b128 at the 2-cycle
//    bank floor); layer-1 weights streamed via a 2x48KB LDS double buffer
//    (one cooperative copy per block instead of 8 per-wave loads)
//  - staging uses GLOBAL_LOAD_ASYNC_TO_LDS (ASYNCcnt) when the builtin is
//    available, register-staged global->ds copy otherwise
//  - ONE atomic grid barrier per timestep (ping-pong h buffers make the
//    second barrier redundant)
// ---------------------------------------------------------------------------

typedef __attribute__((ext_vector_type(8)))  float  v8f;
typedef __attribute__((ext_vector_type(8)))  __bf16 v8bf;
typedef __attribute__((ext_vector_type(16))) __bf16 v16bf;

typedef int v4i __attribute__((vector_size(16)));
typedef __attribute__((address_space(1))) v4i* as1_v4i_p;
typedef __attribute__((address_space(3))) v4i* as3_v4i_p;

#if __has_builtin(__builtin_amdgcn_global_load_async_to_lds_b128)
#define HAS_ASYNC_LDS 1
#else
#define HAS_ASYNC_LDS 0
#endif

#define T_STEPS 1024
#define BATCH   128
#define IN_DIM  6
#define HID     1536
#define OUT_DIM 5
#define GATES   (4 * HID)      // 6144
#define NJT     (HID / 16)     // 96 column tiles -> grid size
#define KSTEP   32             // K per WMMA
#define NKI     (HID / KSTEP)  // 48 k-iterations
#define NTILE   (NKI * 4)      // 192 cached B tiles (layer 0)

#define CH_KK   6                        // k-iters per layer-1 stream chunk
#define NCH     (NKI / CH_KK)            // 8 chunks
#define CH_TILES (CH_KK * 8)             // 48 tiles per chunk
#define CH_ELEMS (CH_TILES * 512)        // 24576 bf16 = 48 KB
#define STG_PER_THREAD ((CH_TILES * 64) / 256)  // 12 x 16B chunks per thread

#define SMEM_L0_ELEMS (NTILE * 512)      // 98304 bf16 = 192 KB
#define SMEM_TOTAL    (SMEM_L0_ELEMS + 2 * CH_ELEMS)   // 147456 bf16 = 288 KB
#define LDS_BYTES     (SMEM_TOTAL * 2)

// ------------------------- workspace layout (bytes) ------------------------
static constexpr size_t SZ_HBUF = 2ull * BATCH * HID * 2;          // ping-pong bf16
static constexpr size_t SZ_WBF  = (size_t)GATES * HID * 2;         // bf16 weight
static constexpr size_t O_BAR   = 0;                               // 2 x u32
static constexpr size_t O_H0    = 256;
static constexpr size_t O_H1    = O_H0 + SZ_HBUF;
static constexpr size_t O_WHH0  = O_H1 + SZ_HBUF;
static constexpr size_t O_WIH1  = O_WHH0 + SZ_WBF;
static constexpr size_t O_WHH1  = O_WIH1 + SZ_WBF;
static constexpr size_t O_B0    = O_WHH1 + SZ_WBF;
static constexpr size_t O_B1    = O_B0 + (size_t)GATES * 4;

// ------------------------------ device helpers -----------------------------
__device__ __forceinline__ float sigmoidf_(float x) {
    return 1.0f / (1.0f + __expf(-x));
}
__device__ __forceinline__ float tanhf_(float x) {
    float e = __expf(2.0f * x);
    return (e - 1.0f) / (e + 1.0f);
}

__device__ __forceinline__ v8f wmma_bf16(v16bf a, v16bf b, v8f c) {
    // D = A(16x32 bf16) * B(32x16 bf16) + C(16x16 f32)
    return __builtin_amdgcn_wmma_f32_16x16x32_bf16(
        /*neg_a=*/false, a, /*neg_b=*/false, b,
        /*c_mod=*/(short)0, c, /*reuse_a=*/false, /*reuse_b=*/false);
}

// Async 16B global -> LDS copy (no VGPR round trip, tracked by ASYNCcnt).
__device__ __forceinline__ void async_copy16(const __bf16* src, __bf16* ldsdst) {
#if HAS_ASYNC_LDS
    __builtin_amdgcn_global_load_async_to_lds_b128(
        (as1_v4i_p)(uintptr_t)src,
        (as3_v4i_p)(unsigned)(uintptr_t)ldsdst,
        /*imm offset=*/0, /*cpol=*/0);
#else
    (void)src; (void)ldsdst;
#endif
}

__device__ __forceinline__ void wait_async_lds() {
#if HAS_ASYNC_LDS
#if __has_builtin(__builtin_amdgcn_s_wait_asynccnt)
    __builtin_amdgcn_s_wait_asynccnt(0);
#else
    asm volatile("s_wait_asynccnt 0x0" ::: "memory");
#endif
#endif
}

// A fragment: 16x32 tile of row-major [BATCH x HID] bf16 matrix.
// ISA layout: lanes 0-15 (M=lane) hold K {0..7, 16..23}; lanes 16-31 hold
// K {8..15, 24..31}.  Two contiguous 16B loads per lane.
__device__ __forceinline__ v16bf load_a(const __bf16* __restrict__ h,
                                        int row, int kk, int hi) {
    const __bf16* p = h + (size_t)row * HID + kk + hi * 8;
    v8bf lo = *(const v8bf*)(p);
    v8bf hh = *(const v8bf*)(p + 16);
    return __builtin_shufflevector(lo, hh, 0, 1, 2, 3, 4, 5, 6, 7,
                                           8, 9, 10, 11, 12, 13, 14, 15);
}

// B fragment from LDS.  Tile layout: element offset
//   tile*512 + half*256 + lane*8   (16B chunk per (lane,half)).
// Each ds_load_b128: 32 lanes x disjoint-start 16B runs, banks collide
// exactly 2-way (lane vs lane+16) -> the 2-cycle LDS floor.
__device__ __forceinline__ v16bf load_b_lds(const __bf16* smem,
                                            int tile, int lane) {
    const __bf16* p = smem + (size_t)tile * 512 + lane * 8;
    v8bf lo = *(const v8bf*)(p);
    v8bf hh = *(const v8bf*)(p + 256);
    return __builtin_shufflevector(lo, hh, 0, 1, 2, 3, 4, 5, 6, 7,
                                           8, 9, 10, 11, 12, 13, 14, 15);
}

__device__ __forceinline__ void grid_barrier(unsigned* bar, unsigned nb) {
    __threadfence();          // every thread orders its global stores
    __syncthreads();
    if (threadIdx.x == 0) {
        unsigned* cnt = bar;
        unsigned* gen = bar + 1;
        unsigned g = atomicAdd(gen, 0u);       // read current generation
        if (atomicAdd(cnt, 1u) == nb - 1) {
            atomicExch(cnt, 0u);
            __threadfence();
            atomicAdd(gen, 1u);                // release
        } else {
            while (atomicAdd(gen, 0u) == g) {
                __builtin_amdgcn_s_sleep(2);
            }
        }
        __threadfence();
    }
    __syncthreads();
}

// ------------------------------- prep kernels ------------------------------
__global__ void cvt_f32_bf16(const float* __restrict__ in,
                             __bf16* __restrict__ out, int n) {
    int i = blockIdx.x * blockDim.x + threadIdx.x;
    int stride = gridDim.x * blockDim.x;
    for (; i < n; i += stride) out[i] = (__bf16)in[i];
}

__global__ void bias_sum(const float* __restrict__ a,
                         const float* __restrict__ b,
                         float* __restrict__ o, int n) {
    int i = blockIdx.x * blockDim.x + threadIdx.x;
    if (i < n) o[i] = a[i] + b[i];
}

__global__ void init_state(__bf16* __restrict__ h0, __bf16* __restrict__ h1,
                           unsigned* __restrict__ bar, int n) {
    int i = blockIdx.x * blockDim.x + threadIdx.x;
    int stride = gridDim.x * blockDim.x;
    for (int k = i; k < n; k += stride) { h0[k] = (__bf16)0.0f; h1[k] = (__bf16)0.0f; }
    if (i == 0) { bar[0] = 0u; bar[1] = 0u; }
}

// ---------------------------- persistent LSTM ------------------------------
__global__ void __launch_bounds__(256, 1)
lstm_persistent(const float* __restrict__ x,        // [T, B, IN] f32
                const float* __restrict__ wih0,     // [GATES, IN] f32
                const __bf16* __restrict__ whh0,    // [GATES, HID] bf16
                const __bf16* __restrict__ wih1,    // [GATES, HID] bf16
                const __bf16* __restrict__ whh1,    // [GATES, HID] bf16
                const float* __restrict__ b0,       // [GATES] f32 (fused)
                const float* __restrict__ b1,       // [GATES] f32 (fused)
                __bf16* __restrict__ h0buf,         // 2 x [B, HID] bf16
                __bf16* __restrict__ h1buf,         // 2 x [B, HID] bf16
                unsigned* __restrict__ bar) {
    extern __shared__ __bf16 smem[];     // 192KB W_hh0 cache + 2x48KB stream

    const int wave = threadIdx.x >> 5;   // 0..7  -> row tile (16 batch rows)
    const int lane = threadIdx.x & 31;
    const int n    = lane & 15;          // output column within tile
    const int hi   = lane >> 4;          // lane half (K split)
    const int jb   = blockIdx.x * 16;    // j-tile base within HID
    const int rb   = wave * 16;          // batch-row base
    const int arow = rb + n;             // A-fragment source row

    // ---- one-time: stage this block's W_hh0 slice into LDS (192 KB) ----
    // chunk c: tile = c>>6 (= kkIdx*4+g), half = (c>>5)&1, ln = c&31.
    for (int c = threadIdx.x; c < NTILE * 64; c += blockDim.x) {
        int tile = c >> 6, sub = c & 63;
        int half = sub >> 5, ln = sub & 31;
        int g = tile & 3, kkIdx = tile >> 2;
        int wrow = g * HID + jb + (ln & 15);
        int k = kkIdx * KSTEP + (ln >> 4) * 8 + half * 16;
        v8bf v = *(const v8bf*)(whh0 + (size_t)wrow * HID + k);
        *(v8bf*)(&smem[(size_t)c * 8]) = v;
    }

    // Per-thread layer-1 staging descriptors (time/chunk invariant).
    // Linear 16B chunk cc within a chunk buffer:
    //   tile tl = cc>>6 (= kkL*8 + g*2 + w), half = (cc>>5)&1, ln = cc&31.
    const __bf16* stg_src[STG_PER_THREAD];
    int stg_lds[STG_PER_THREAD];
    #pragma unroll
    for (int i = 0; i < STG_PER_THREAD; ++i) {
        int cc = threadIdx.x + i * 256;
        int tl = cc >> 6, sub = cc & 63;
        int half = sub >> 5, ln = sub & 31;
        int kkL = tl >> 3, g = (tl >> 1) & 3, w = tl & 1;
        const __bf16* mat = w ? whh1 : wih1;
        stg_src[i] = mat + (size_t)(g * HID + jb + (ln & 15)) * HID
                         + kkL * KSTEP + (ln >> 4) * 8 + half * 16;
        stg_lds[i] = SMEM_L0_ELEMS + cc * 8;
    }

    // Time-invariant per-lane constants: fused biases and W_ih0 rows (K=6).
    float bias0[4], bias1[4], w0[4][IN_DIM];
    #pragma unroll
    for (int g = 0; g < 4; ++g) {
        bias0[g] = b0[g * HID + jb + n];
        bias1[g] = b1[g * HID + jb + n];
        const float* wr = wih0 + (size_t)(g * HID + jb + n) * IN_DIM;
        #pragma unroll
        for (int k = 0; k < IN_DIM; ++k) w0[g][k] = wr[k];
    }

    // Cell state lives in VGPRs for the whole sequence.
    float c0[8], c1[8];
    #pragma unroll
    for (int r = 0; r < 8; ++r) { c0[r] = 0.0f; c1[r] = 0.0f; }

    __syncthreads();                     // LDS weight cache ready

    const unsigned nblocks = gridDim.x;

    for (int t = 0; t < T_STEPS; ++t) {
        const int ri = t & 1, wi = ri ^ 1;
        const __bf16* h0r = h0buf + (size_t)ri * BATCH * HID;
        const __bf16* h1r = h1buf + (size_t)ri * BATCH * HID;
        __bf16* h0w = h0buf + (size_t)wi * BATCH * HID;
        __bf16* h1w = h1buf + (size_t)wi * BATCH * HID;

        // ---- layer 0: gates = b0 + x_t W_ih0^T + h0 W_hh0^T ----
        // A from global (VMEM path), B from LDS cache (DS path) in parallel.
        v8f acc[4];
        #pragma unroll
        for (int g = 0; g < 4; ++g)
            #pragma unroll
            for (int r = 0; r < 8; ++r) acc[g][r] = bias0[g];

        for (int kk = 0; kk < HID; kk += KSTEP) {
            const int tbase = (kk >> 5) * 4;          // kkIdx*4
            v16bf a = load_a(h0r, arow, kk, hi);
            #pragma unroll
            for (int g = 0; g < 4; ++g) {
                v16bf bt = load_b_lds(smem, tbase + g, lane);
                acc[g] = wmma_bf16(a, bt, acc[g]);
            }
        }

        // prefetch next step's x rows into WGP-level cache
        if (t + 1 < T_STEPS) {
            __builtin_prefetch(x + ((size_t)(t + 1) * BATCH + rb + hi * 8) * IN_DIM, 0, 3);
        }

        #pragma unroll
        for (int r = 0; r < 8; ++r) {
            const int brow = rb + hi * 8 + r;       // C-layout: M = r (+8 hi)
            const float* xr = x + ((size_t)t * BATCH + brow) * IN_DIM;
            float gi = acc[0][r], gf = acc[1][r], gg = acc[2][r], go = acc[3][r];
            #pragma unroll
            for (int k = 0; k < IN_DIM; ++k) {
                float xv = xr[k];
                gi += xv * w0[0][k]; gf += xv * w0[1][k];
                gg += xv * w0[2][k]; go += xv * w0[3][k];
            }
            float cn = sigmoidf_(gf) * c0[r] + sigmoidf_(gi) * tanhf_(gg);
            float hn = sigmoidf_(go) * tanhf_(cn);
            c0[r] = cn;
            h0w[(size_t)brow * HID + jb + n] = (__bf16)hn;
        }

        // ---- stage chunk 0 of layer-1 B; latency hides under the barrier --
#if HAS_ASYNC_LDS
        #pragma unroll
        for (int i = 0; i < STG_PER_THREAD; ++i)
            async_copy16(stg_src[i], &smem[stg_lds[i]]);

        grid_barrier(bar, nblocks);   // the ONLY device-wide sync per step

        wait_async_lds();
        __syncthreads();
#else
        v8bf S[STG_PER_THREAD];
        #pragma unroll
        for (int i = 0; i < STG_PER_THREAD; ++i)
            S[i] = *(const v8bf*)(stg_src[i]);

        grid_barrier(bar, nblocks);   // the ONLY device-wide sync per step

        #pragma unroll
        for (int i = 0; i < STG_PER_THREAD; ++i)
            *(v8bf*)(&smem[stg_lds[i]]) = S[i];
        __syncthreads();
#endif

        // ---- layer 1: gates = b1 + h0_new W_ih1^T + h1_old W_hh1^T ----
        // B tiles double-buffered through LDS; A tiles stream on VMEM path.
        #pragma unroll
        for (int g = 0; g < 4; ++g)
            #pragma unroll
            for (int r = 0; r < 8; ++r) acc[g][r] = bias1[g];

        for (int c = 0; c < NCH; ++c) {
            if (c + 1 < NCH) {              // prefetch next chunk
                const int koff = (c + 1) * CH_KK * KSTEP;
                const int boff = ((c + 1) & 1) * CH_ELEMS;
#if HAS_ASYNC_LDS
                #pragma unroll
                for (int i = 0; i < STG_PER_THREAD; ++i)
                    async_copy16(stg_src[i] + koff, &smem[stg_lds[i] + boff]);
#else
                #pragma unroll
                for (int i = 0; i < STG_PER_THREAD; ++i)
                    S[i] = *(const v8bf*)(stg_src[i] + koff);
#endif
            }
            const __bf16* bufp = smem + SMEM_L0_ELEMS + (c & 1) * CH_ELEMS;
            #pragma unroll
            for (int kkL = 0; kkL < CH_KK; ++kkL) {
                const int kk = (c * CH_KK + kkL) * KSTEP;
                v16bf a0 = load_a(h0w, arow, kk, hi);   // new h0
                v16bf a1 = load_a(h1r, arow, kk, hi);   // previous h1
                const int tb = kkL * 8;
                #pragma unroll
                for (int g = 0; g < 4; ++g) {
                    v16bf bI = load_b_lds(bufp, tb + g * 2, lane);
                    acc[g] = wmma_bf16(a0, bI, acc[g]);
                    v16bf bH = load_b_lds(bufp, tb + g * 2 + 1, lane);
                    acc[g] = wmma_bf16(a1, bH, acc[g]);
                }
            }
            if (c + 1 < NCH) {
#if HAS_ASYNC_LDS
                wait_async_lds();
#else
                const int boff = ((c + 1) & 1) * CH_ELEMS;
                #pragma unroll
                for (int i = 0; i < STG_PER_THREAD; ++i)
                    *(v8bf*)(&smem[stg_lds[i] + boff]) = S[i];
#endif
                __syncthreads();
            }
        }

        #pragma unroll
        for (int r = 0; r < 8; ++r) {
            const int brow = rb + hi * 8 + r;
            float gi = acc[0][r], gf = acc[1][r], gg = acc[2][r], go = acc[3][r];
            float cn = sigmoidf_(gf) * c1[r] + sigmoidf_(gi) * tanhf_(gg);
            float hn = sigmoidf_(go) * tanhf_(cn);
            c1[r] = cn;
            h1w[(size_t)brow * HID + jb + n] = (__bf16)hn;
        }
        // No second grid barrier: ping-pong buffers + next step's barrier
        // cover every RAW/WAR edge.
    }
}

// ----------------------------- final linear --------------------------------
__global__ void final_linear(const __bf16* __restrict__ h1,   // [B, HID] bf16
                             const float* __restrict__ wlin,  // [OUT, HID]
                             const float* __restrict__ blin,  // [OUT]
                             float* __restrict__ out) {       // [B, OUT]
    int idx = blockIdx.x * blockDim.x + threadIdx.x;
    if (idx >= BATCH * OUT_DIM) return;
    int b = idx / OUT_DIM, o = idx % OUT_DIM;
    const __bf16* hr = h1 + (size_t)b * HID;
    const float* wr = wlin + (size_t)o * HID;
    float s = blin[o];
    for (int k = 0; k < HID; ++k) s += (float)hr[k] * wr[k];
    out[idx] = s;
}

// ------------------------------- launcher ----------------------------------
extern "C" void kernel_launch(void* const* d_in, const int* in_sizes, int n_in,
                              void* d_out, int out_size, void* d_ws, size_t ws_size,
                              hipStream_t stream) {
    (void)in_sizes; (void)n_in; (void)out_size; (void)ws_size;

    const float* x     = (const float*)d_in[0];   // [T, B, IN]
    const float* wih0  = (const float*)d_in[1];   // [4H, IN]
    const float* whh0f = (const float*)d_in[2];   // [4H, H]
    const float* bih0  = (const float*)d_in[3];
    const float* bhh0  = (const float*)d_in[4];
    const float* wih1f = (const float*)d_in[5];   // [4H, H]
    const float* whh1f = (const float*)d_in[6];   // [4H, H]
    const float* bih1  = (const float*)d_in[7];
    const float* bhh1  = (const float*)d_in[8];
    const float* wlin  = (const float*)d_in[9];   // [OUT, H]
    const float* blin  = (const float*)d_in[10];
    float* out = (float*)d_out;

    char* ws = (char*)d_ws;
    unsigned* bar   = (unsigned*)(ws + O_BAR);
    __bf16*   h0b   = (__bf16*)(ws + O_H0);
    __bf16*   h1b   = (__bf16*)(ws + O_H1);
    __bf16*   whh0  = (__bf16*)(ws + O_WHH0);
    __bf16*   wih1  = (__bf16*)(ws + O_WIH1);
    __bf16*   whh1  = (__bf16*)(ws + O_WHH1);
    float*    b0    = (float*)(ws + O_B0);
    float*    b1    = (float*)(ws + O_B1);

    const int welem = GATES * HID;                 // 9,437,184
    cvt_f32_bf16<<<2048, 256, 0, stream>>>(whh0f, whh0, welem);
    cvt_f32_bf16<<<2048, 256, 0, stream>>>(wih1f, wih1, welem);
    cvt_f32_bf16<<<2048, 256, 0, stream>>>(whh1f, whh1, welem);
    bias_sum<<<(GATES + 255) / 256, 256, 0, stream>>>(bih0, bhh0, b0, GATES);
    bias_sum<<<(GATES + 255) / 256, 256, 0, stream>>>(bih1, bhh1, b1, GATES);
    init_state<<<256, 256, 0, stream>>>(h0b, h1b, bar, 2 * BATCH * HID);

    lstm_persistent<<<NJT, 256, LDS_BYTES, stream>>>(x, wih0, whh0, wih1, whh1,
                                                     b0, b1, h0b, h1b, bar);

    // t = 1023 writes ping-pong slot 0 -> final h1 lives at offset 0.
    final_linear<<<(BATCH * OUT_DIM + 127) / 128, 128, 0, stream>>>(
        h1b, wlin, blin, out);
}
